// SigmaMoE_1666447311383
// MI455X (gfx1250) — matrix operand order repaired and checked by reading, not compiled, not run on hardware
//
#include <hip/hip_runtime.h>
#include <hip/hip_bf16.h>
#include <math.h>

#define DMODEL 1024
#define NEXP   16
#define HSIZE  256
#define NTOK   8192
#define TM     32        // tokens per FFN block

// LDS strides
#define XS 1032          // x tile row stride in bf16 (dword stride 516 -> bank step 4)
#define HS 264           // h tile row stride in bf16 (dword stride 132 -> bank step 4)
#define CS 36            // slab column stride in bf16 (72B: distinct start banks, 8B aligned)

#define SLAB_HALFS (256 * 32)   // one staged weight slab: 256 cols x 32 K, bf16 packed

typedef __attribute__((ext_vector_type(16))) __bf16 v16bf;
typedef __attribute__((ext_vector_type(8)))  float  v8f;

__device__ __forceinline__ unsigned pack_bf16(float a, float b) {
  union { __bf16 h[2]; unsigned u; } cv;
  cv.h[0] = (__bf16)a; cv.h[1] = (__bf16)b;
  return cv.u;
}

// A-matrix (16x32 bf16): element i=2v+p in lane (m = lane&15,
// hi = lane>>4) holds K = 16*hi + 2*v + p. Pairs contiguous ->
// b32/b128-mergeable LDS reads. `p` points at row start + k0.
__device__ __forceinline__ v16bf load_a(const __bf16* p, int hi) {
  v16bf r;
#pragma unroll
  for (int v = 0; v < 8; ++v) {
    int koff = (v << 1) + (hi << 4);
    union { unsigned u; __bf16 h[2]; } cv;
    cv.u = *(const unsigned*)(p + koff);
    r[2 * v]     = cv.h[0];
    r[2 * v + 1] = cv.h[1];
  }
  return r;
}

// B-matrix (32x16 bf16): lanes 0-15 hold K=0..15, lanes 16-31 hold K=16..31.
// Weights are pre-converted to [n][kk] column-major bf16, so the operand is a
// contiguous 32B run at &col[16*hi]: pure wide LDS loads, no assembly VALU.
__device__ __forceinline__ v16bf load_b(const __bf16* col, int hi) {
  union { v16bf v; unsigned long long q[4]; } u;
  const unsigned long long* p64 = (const unsigned long long*)(col + (hi << 4));
  u.q[0] = p64[0]; u.q[1] = p64[1]; u.q[2] = p64[2]; u.q[3] = p64[3];
  return u.v;
}

__device__ __forceinline__ void wait_async0() {
#if __has_builtin(__builtin_amdgcn_s_wait_asynccnt)
  __builtin_amdgcn_s_wait_asynccnt(0);
#else
  asm volatile("s_wait_asynccnt 0x0" ::: "memory");
#endif
}

// Async-stage one 16KB bf16 weight slab (packed [n][kk], 64B per column) into
// LDS with column stride CS (72B). 64 b64 transfers split across 8 waves.
__device__ __forceinline__ void stage_slab(unsigned lds_off, const __bf16* gslab,
                                           int wave, int lane) {
  const char* gs = (const char*)gslab + lane * 8;
  unsigned     ls = lds_off + (unsigned)((lane >> 3) * (CS * 2)) + (unsigned)((lane & 7) * 8);
#pragma unroll
  for (int t = 0; t < 8; ++t) {
    int i = wave * 8 + t;                       // 0..63 chunk of 256B src / 4 cols
    const char* g0 = gs + i * 256;
    unsigned    l0 = ls + (unsigned)(i * 4 * CS * 2);
    asm volatile("global_load_async_to_lds_b64 %0, %1, off"
                 :: "v"(l0), "v"(g0) : "memory");
  }
}

// ---------------- Weight pre-convert: fp32 -> bf16, WMMA-B-operand order ----
// keys  [e][d][n]   -> kb[(e*32+kt)*8192 + n*32 + kk]   (d = kt*32+kk)
// values[e][k][c]   -> vb[((e*4+q)*8+kt)*8192 + n*32 + kk]  (k = kt*32+kk, c = q*256+n)
__global__ __launch_bounds__(256, 2)
void convert_weights(const float* __restrict__ keys, const float* __restrict__ values,
                     __bf16* __restrict__ kb, __bf16* __restrict__ vb) {
  __shared__ float t[32 * 257];
  const int b   = blockIdx.x;
  const int tid = threadIdx.x;
  const float* src;
  __bf16* dst;
  size_t rstride;
  if (b < 512) {                    // keys: b = e*32 + kt
    int e = b >> 5, kt = b & 31;
    src = keys + ((size_t)e * DMODEL + (size_t)kt * 32) * HSIZE;
    dst = kb + (size_t)b * SLAB_HALFS;
    rstride = HSIZE;
  } else {                          // values: b-512 = (e*4+q)*8 + kt
    int bb = b - 512;
    int e = bb >> 5, q = (bb >> 3) & 3, kt = bb & 7;
    src = values + ((size_t)e * HSIZE + (size_t)kt * 32) * DMODEL + (size_t)q * 256;
    dst = vb + (size_t)bb * SLAB_HALFS;
    rstride = DMODEL;
  }
  for (int i = tid; i < 8192; i += 256) {
    int d = i >> 8, n = i & 255;                 // coalesced reads
    t[d * 257 + n] = src[(size_t)d * rstride + n];
  }
  __syncthreads();
  for (int i = tid; i < 8192; i += 256) {
    int n = i >> 5, kk = i & 31;                 // coalesced packed writes
    dst[i] = (__bf16)t[kk * 257 + n];
  }
}

// ---------------- Routing: sigmoid gate + top-4 + per-expert compaction -----
__global__ __launch_bounds__(256, 4)
void moe_route(const float* __restrict__ x, const float* __restrict__ sel,
               const float* __restrict__ route_scale,
               int* __restrict__ counts, int* __restrict__ toks,
               float* __restrict__ wvals) {
  __shared__ float s_sel[NEXP * DMODEL];   // 64 KB
  const int tid = threadIdx.x;
  for (int i = tid; i < NEXP * DMODEL; i += 256) s_sel[i] = sel[i];
  __syncthreads();

  const int wave = tid >> 5;
  const int lane = tid & 31;
  const int tok  = blockIdx.x * 8 + wave;

  float acc[NEXP];
#pragma unroll
  for (int e = 0; e < NEXP; ++e) acc[e] = 0.f;

  const float* xp = x + (size_t)tok * DMODEL;
  for (int d0 = 0; d0 < DMODEL; d0 += 32) {
    float xv = xp[d0 + lane];
#pragma unroll
    for (int e = 0; e < NEXP; ++e) acc[e] += xv * s_sel[e * DMODEL + d0 + lane];
  }
#pragma unroll
  for (int e = 0; e < NEXP; ++e) {
#pragma unroll
    for (int off = 16; off > 0; off >>= 1)
      acc[e] += __shfl_xor(acc[e], off, 32);
  }
  float p[NEXP];
#pragma unroll
  for (int e = 0; e < NEXP; ++e) p[e] = 1.f / (1.f + __expf(-acc[e]));

  unsigned taken = 0; float wsum = 0.f;
  int idx[4]; float val[4];
#pragma unroll
  for (int k = 0; k < 4; ++k) {
    float best = -1.f; int bi = 0;
#pragma unroll
    for (int e = 0; e < NEXP; ++e) {
      bool ok = (((taken >> e) & 1u) == 0u) && (p[e] > best);
      if (ok) { best = p[e]; bi = e; }
    }
    taken |= 1u << bi;
    idx[k] = bi; val[k] = best; wsum += best;
  }
  const float scale = route_scale[0] / fmaxf(wsum, 1e-9f);

  if (lane < 4) {
    int myi = idx[0]; float myv = val[0];
    if (lane == 1) { myi = idx[1]; myv = val[1]; }
    else if (lane == 2) { myi = idx[2]; myv = val[2]; }
    else if (lane == 3) { myi = idx[3]; myv = val[3]; }
    int pos = atomicAdd(&counts[myi], 1);
    toks[myi * NTOK + pos]  = tok;
    wvals[myi * NTOK + pos] = myv * scale;
  }
}

// ---------------- Expert FFN: out += w * (relu(Xg K_e) V_e), bf16 WMMA ------
__global__ __launch_bounds__(256, 1)
void moe_ffn(const float* __restrict__ x, const __bf16* __restrict__ kb,
             const __bf16* __restrict__ vb, const int* __restrict__ counts,
             const int* __restrict__ toks, const float* __restrict__ wvals,
             float* __restrict__ out) {
  __shared__ __align__(16) __bf16 lds_x[TM * XS];       // gathered activations
  __shared__ __align__(16) __bf16 lds_h[TM * HS];       // relu hidden
  __shared__ __align__(16) __bf16 lds_s[2][256 * CS];   // async weight slabs
  __shared__ int    lds_tok[TM];
  __shared__ float  lds_w[TM];

  const int e   = blockIdx.y;
  const int m0  = blockIdx.x * TM;
  const int cnt = counts[e];
  if (m0 >= cnt) return;               // uniform exit

  const int tid  = threadIdx.x;
  const int wave = tid >> 5;
  const int lane = tid & 31;
  const int n16  = lane & 15;
  const int hi   = lane >> 4;
  const int mt   = wave >> 2;          // 0..1  (M-tile of 16)
  const int ng   = wave & 3;           // 0..3  (N-group)

  if (tid < TM) {
    int g = m0 + tid;
    if (g < cnt) { lds_tok[tid] = toks[e * NTOK + g]; lds_w[tid] = wvals[e * NTOK + g]; }
    else         { lds_tok[tid] = 0;                  lds_w[tid] = 0.f; }  // pad: weight 0
  }
  __syncthreads();

  // gather X rows -> LDS bf16 (packed pairs, coalesced global reads)
  for (int i = tid; i < TM * (DMODEL / 2); i += 256) {
    int r  = i >> 9;                 // DMODEL/2 == 512
    int dp = (i & 511) << 1;
    const float* src = x + (size_t)lds_tok[r] * DMODEL + dp;
    *(unsigned*)&lds_x[r * XS + dp] = pack_bf16(src[0], src[1]);
  }
  __syncthreads();

  const v8f vzero = {0.f, 0.f, 0.f, 0.f, 0.f, 0.f, 0.f, 0.f};
  unsigned sboff[2];
  sboff[0] = (unsigned)(uintptr_t)&lds_s[0][0];
  sboff[1] = (unsigned)(uintptr_t)&lds_s[1][0];

  // ---- GEMM1: h = relu(Xg[32,1024] * K_e[1024,256]) -----------------------
  v8f acc1[4];
#pragma unroll
  for (int j = 0; j < 4; ++j) acc1[j] = vzero;

  const __bf16* kslabs = kb + (size_t)e * 32 * SLAB_HALFS;
  stage_slab(sboff[0], kslabs, wave, lane);                       // prime
  for (int kt = 0; kt < DMODEL / 32; ++kt) {
    wait_async0();
    __syncthreads();                                              // slab[kt&1] ready everywhere
    if (kt + 1 < DMODEL / 32)
      stage_slab(sboff[(kt + 1) & 1], kslabs + (size_t)(kt + 1) * SLAB_HALFS, wave, lane);
    const __bf16* slab = &lds_s[kt & 1][0];
    v16bf a = load_a(&lds_x[(mt * 16 + n16) * XS + kt * 32], hi);
#pragma unroll
    for (int j = 0; j < 4; ++j) {
      int nt = ng * 4 + j;
      v16bf b = load_b(&slab[(nt * 16 + n16) * CS], hi);
      acc1[j] = __builtin_amdgcn_wmma_f32_16x16x32_bf16(
          false, a, false, b, (short)0, acc1[j], false, false);
    }
  }

  // relu + convert -> lds_h (C/D layout: VGPR r -> row r + 8*hi)
#pragma unroll
  for (int j = 0; j < 4; ++j) {
    int nt = ng * 4 + j;
#pragma unroll
    for (int r = 0; r < 8; ++r) {
      float v = acc1[j][r];
      v = v > 0.f ? v : 0.f;
      lds_h[(mt * 16 + r + hi * 8) * HS + nt * 16 + n16] = (__bf16)v;
    }
  }

  int   rtok[8];
  float rw[8];
#pragma unroll
  for (int r = 0; r < 8; ++r) {
    int m = mt * 16 + r + hi * 8;
    rtok[r] = lds_tok[m];
    rw[r]   = lds_w[m];
  }

  // ---- GEMM2: y = h[32,256] * V_e[256,1024], four 256-col quarters --------
  const __bf16* vslabs = vb + (size_t)e * 32 * SLAB_HALFS;   // 4 quarters x 8 ksteps
  stage_slab(sboff[0], vslabs, wave, lane);                  // prime s=0
  for (int q = 0; q < 4; ++q) {
    v8f acc2[4];
#pragma unroll
    for (int j = 0; j < 4; ++j) acc2[j] = vzero;

    for (int kt = 0; kt < HSIZE / 32; ++kt) {
      const int s = q * 8 + kt;
      wait_async0();
      __syncthreads();                 // slab[s&1] ready; lds_h visible (first iter)
      const int ns = s + 1;
      if (ns < 32)
        stage_slab(sboff[ns & 1], vslabs + (size_t)ns * SLAB_HALFS, wave, lane);
      const __bf16* slab = &lds_s[s & 1][0];
      v16bf a = load_a(&lds_h[(mt * 16 + n16) * HS + kt * 32], hi);
#pragma unroll
      for (int j = 0; j < 4; ++j) {
        int nt = ng * 4 + j;
        v16bf b = load_b(&slab[(nt * 16 + n16) * CS], hi);
        acc2[j] = __builtin_amdgcn_wmma_f32_16x16x32_bf16(
            false, a, false, b, (short)0, acc2[j], false, false);
      }
    }
    // scale by routing weight and combine across the 4 experts
#pragma unroll
    for (int j = 0; j < 4; ++j) {
      int col = q * 256 + (ng * 4 + j) * 16 + n16;
#pragma unroll
      for (int r = 0; r < 8; ++r) {
        atomicAdd(&out[(size_t)rtok[r] * DMODEL + col], acc2[j][r] * rw[r]);
      }
    }
  }
}

// ---------------------------------------------------------------------------
extern "C" void kernel_launch(void* const* d_in, const int* in_sizes, int n_in,
                              void* d_out, int out_size, void* d_ws, size_t ws_size,
                              hipStream_t stream) {
  const float* x      = (const float*)d_in[0];   // [4,2048,1024]
  const float* sel    = (const float*)d_in[1];   // [16,1024]
  const float* keys   = (const float*)d_in[2];   // [16,1024,256]
  const float* values = (const float*)d_in[3];   // [16,256,1024]
  const float* rscale = (const float*)d_in[4];   // [1]
  float* out = (float*)d_out;

  // workspace: bf16 keys (8.39MB) | bf16 values (8.39MB) | counts | lists
  const size_t WBYTES = (size_t)512 * SLAB_HALFS * sizeof(__bf16);  // 8,388,608
  __bf16* kb     = (__bf16*)d_ws;
  __bf16* vb     = (__bf16*)((char*)d_ws + WBYTES);
  char*   meta   = (char*)d_ws + 2 * WBYTES;
  int*    counts = (int*)meta;
  int*    toks   = (int*)(meta + 256);
  float*  wvals  = (float*)(meta + 256 + (size_t)NEXP * NTOK * sizeof(int));

  hipMemsetAsync(meta, 0, 256, stream);                               // counters
  hipMemsetAsync(d_out, 0, (size_t)out_size * sizeof(float), stream); // accum target

  convert_weights<<<1024, 256, 0, stream>>>(keys, values, kb, vb);
  moe_route<<<NTOK / 8, 256, 0, stream>>>(x, sel, rscale, counts, toks, wvals);
  moe_ffn<<<dim3(NTOK / TM, NEXP), 256, 0, stream>>>(x, kb, vb,
                                                     counts, toks, wvals, out);
}
